// FusedMoELinear_13400297963952
// MI455X (gfx1250) — matrix operand order repaired and checked by reading, not compiled, not run on hardware
//
#include <hip/hip_runtime.h>
#include <hip/hip_bf16.h>

// Problem constants (from reference)
#define EXP  8
#define KK   2048
#define NN   2048
#define TT   1024
#define TOPK 2
#define NG   16           // K / GROUP_SIZE
#define LISTSTRIDE 2048   // max pairs per expert = T*TOPK

// GEMM tiling
#define BLOCK_M 256       // 8 waves x 2 m-frags x 16 rows
#define BLOCK_N 64        // 4 n-frags of 16
#define KSTEP   32
#define NITER   (KK / KSTEP)

typedef __attribute__((ext_vector_type(16))) _Float16 v16h;
typedef __attribute__((ext_vector_type(8)))  float    v8f;

struct Halves { uint4 lo; uint4 hi; };

static __device__ inline v16h load_frag(const _Float16* p0, const _Float16* p1) {
    Halves h;
    h.lo = *reinterpret_cast<const uint4*>(p0);
    h.hi = *reinterpret_cast<const uint4*>(p1);
    return __builtin_bit_cast(v16h, h);
}

// ---------------------------------------------------------------- init
__global__ __launch_bounds__(256) void moe_init_kernel(float* __restrict__ out,
                                                       int* __restrict__ cnt) {
    long long i = (long long)blockIdx.x * blockDim.x + threadIdx.x;
    if (i < (long long)TT * NN) out[i] = 0.0f;
    if (i < EXP) cnt[i] = 0;
}

// ---------------------------------------------------------------- route
__global__ __launch_bounds__(256) void moe_route_kernel(const int* __restrict__ ids,
                                                        const float* __restrict__ tw,
                                                        int* __restrict__ cnt,
                                                        int* __restrict__ tok,
                                                        float* __restrict__ gate) {
    int p = blockIdx.x * blockDim.x + threadIdx.x;
    if (p >= TT * TOPK) return;
    int e = ids[p];
    int pos = atomicAdd(&cnt[e], 1);
    tok[e * LISTSTRIDE + pos]  = p / TOPK;
    gate[e * LISTSTRIDE + pos] = tw[p];
}

// ---------------------------------------------------------------- bias: dot(x[t], z[e]) per routed pair
__global__ __launch_bounds__(256) void moe_bias_kernel(const float* __restrict__ x,
                                                       const float* __restrict__ zeros,
                                                       const int* __restrict__ tok,
                                                       const int* __restrict__ cnt,
                                                       float* __restrict__ bias) {
    int e    = blockIdx.x >> 8;           // 256 blocks per expert
    int base = (blockIdx.x & 255) * 8;    // 8 waves per block -> 8 pairs
    int wave = threadIdx.x >> 5;
    int lane = threadIdx.x & 31;
    int idx  = base + wave;
    if (idx >= cnt[e]) return;
    int t = tok[e * LISTSTRIDE + idx];
    const float* xr = x + (long long)t * KK;
    const float* zr = zeros + (long long)e * KK * NG;
    float acc = 0.0f;
    for (int k = lane; k < KK; k += 32)
        acc += xr[k] * zr[k * NG + (k >> 7)];      // zeros[e,k,k/128]
    #pragma unroll
    for (int off = 16; off; off >>= 1)
        acc += __shfl_xor(acc, off, 32);
    if (lane == 0) bias[e * LISTSTRIDE + idx] = acc;
}

// ---------------------------------------------------------------- main fused GEMM
// A = x[token rows]        (f16, plain convert)
// B = w_q * s[e,k]         (f16, scale folded into B since scale is per-k only)
// out[t,n] += gate * (A@B + bias)   via atomic f32 adds
__global__ __launch_bounds__(256) void moe_gemm_kernel(const float* __restrict__ x,
                                                       const int*   __restrict__ wq,
                                                       const float* __restrict__ scales,
                                                       const int*   __restrict__ tok,
                                                       const float* __restrict__ gate,
                                                       const float* __restrict__ bias,
                                                       const int*   __restrict__ cnt,
                                                       float*       __restrict__ out) {
    const int e  = blockIdx.z;
    const int mb = blockIdx.y * BLOCK_M;
    const int nb = blockIdx.x * BLOCK_N;
    const int valid = cnt[e] - mb;         // rows of this expert in this m-tile
    if (valid <= 0) return;

    __shared__ _Float16 Ash[2][BLOCK_M][KSTEP];   // 2 x 16 KB
    __shared__ _Float16 Bsh[2][BLOCK_N][KSTEP];   // 2 x 4 KB, [n][k] (transposed)
    __shared__ float    sSc[KK];                  // 8 KB: s[e,k] for whole K

    const int tid  = threadIdx.x;
    const int lane = tid & 31;
    const int wave = tid >> 5;             // 0..7

    // preload full per-expert scale row: s[e,k] = scales[e,k,k/128]
    for (int i = tid; i < KK; i += 256)
        sSc[i] = scales[((long long)e * KK + i) * NG + (i >> 7)];

    // ---- A-gather bookkeeping: thread covers 8 rows (rseg + 32p), float4 of k each
    const int kcol = tid & 7;              // 8 threads * float4 = 32 k
    const int rseg = tid >> 3;             // 0..31
    int   trow[8];
    float rmask[8];
    #pragma unroll
    for (int p = 0; p < 8; ++p) {
        int r = rseg + 32 * p;
        bool ok = (r < valid);
        trow[p]  = ok ? tok[e * LISTSTRIDE + mb + r] : 0;
        rmask[p] = ok ? 1.0f : 0.0f;
    }

    // ---- B loader: one n-column, 8 contiguous k per thread
    const int bn  = tid & 63;
    const int kb8 = (tid >> 6) * 8;        // 0,8,16,24
    const long long wq_base = ((long long)e * KK) * NN + nb + bn;

    v8f acc[2][4];
    const v8f vzero = {0.f, 0.f, 0.f, 0.f, 0.f, 0.f, 0.f, 0.f};
    #pragma unroll
    for (int mi = 0; mi < 2; ++mi)
        #pragma unroll
        for (int nt = 0; nt < 4; ++nt) acc[mi][nt] = vzero;

    // register staging for double buffering
    float4 ax[8];
    int    bw[8];
    float4 bs0, bs1;

    auto load_stage = [&](int kb) {
        const int* bp = wq + wq_base + (long long)(kb + kb8) * NN;
        #pragma unroll
        for (int p = 0; p < 8; ++p) bw[p] = bp[(long long)p * NN];
        #pragma unroll
        for (int p = 0; p < 8; ++p)
            ax[p] = *reinterpret_cast<const float4*>(
                x + (long long)trow[p] * KK + kb + kcol * 4);
        bs0 = *reinterpret_cast<const float4*>(&sSc[kb + kb8]);
        bs1 = *reinterpret_cast<const float4*>(&sSc[kb + kb8 + 4]);
    };

    auto store_stage = [&](int buf) {
        _Float16* bd = &Bsh[buf][bn][kb8];           // 8 contiguous halfs
        bd[0] = (_Float16)((float)bw[0] * bs0.x);
        bd[1] = (_Float16)((float)bw[1] * bs0.y);
        bd[2] = (_Float16)((float)bw[2] * bs0.z);
        bd[3] = (_Float16)((float)bw[3] * bs0.w);
        bd[4] = (_Float16)((float)bw[4] * bs1.x);
        bd[5] = (_Float16)((float)bw[5] * bs1.y);
        bd[6] = (_Float16)((float)bw[6] * bs1.z);
        bd[7] = (_Float16)((float)bw[7] * bs1.w);
        #pragma unroll
        for (int p = 0; p < 8; ++p) {
            int r = rseg + 32 * p;
            float m = rmask[p];
            _Float16* ad = &Ash[buf][r][kcol * 4];   // 4 contiguous halfs
            ad[0] = (_Float16)(ax[p].x * m);
            ad[1] = (_Float16)(ax[p].y * m);
            ad[2] = (_Float16)(ax[p].z * m);
            ad[3] = (_Float16)(ax[p].w * m);
        }
    };

    __syncthreads();                 // scales visible
    load_stage(0);
    store_stage(0);
    __syncthreads();

    const int arow = lane & 15;
    const int ak0  = (lane >> 4) * 8;   // A frag: k 0..7|16..23 (lo lanes), 8..15|24..31 (hi)
    const int bk0  = (lane >> 4) * 8;   // B frag: k 0..7|16..23 (lo lanes), 8..15|24..31 (hi)
    const int bnl  = lane & 15;

    for (int it = 0; it < NITER; ++it) {
        const int cur = it & 1;
        const bool more = (it + 1) < NITER;
        if (more) load_stage((it + 1) * KSTEP);   // global loads overlap WMMA below

        v16h a0 = load_frag(&Ash[cur][wave * 32 + arow][ak0],
                            &Ash[cur][wave * 32 + arow][ak0 + 16]);
        v16h a1 = load_frag(&Ash[cur][wave * 32 + 16 + arow][ak0],
                            &Ash[cur][wave * 32 + 16 + arow][ak0 + 16]);
        #pragma unroll
        for (int nt = 0; nt < 4; ++nt) {
            v16h b = load_frag(&Bsh[cur][nt * 16 + bnl][bk0],
                               &Bsh[cur][nt * 16 + bnl][bk0 + 16]);
            acc[0][nt] = __builtin_amdgcn_wmma_f32_16x16x32_f16(
                false, a0, false, b, (short)0, acc[0][nt], false, false);
            acc[1][nt] = __builtin_amdgcn_wmma_f32_16x16x32_f16(
                false, a1, false, b, (short)0, acc[1][nt], false, false);
        }

        if (more) store_stage(cur ^ 1);
        __syncthreads();
    }

    // ---- epilogue: out[t, n] += gate * (acc + bias)
    // C/D layout: VGPR i -> M = i (lanes 0-15) / i+8 (lanes 16-31); N = lane&15.
    const int laneN  = lane & 15;
    const int rowOff = (lane >> 4) * 8;
    #pragma unroll
    for (int mi = 0; mi < 2; ++mi) {
        #pragma unroll
        for (int i = 0; i < 8; ++i) {
            const int mloc = wave * 32 + mi * 16 + i + rowOff;
            if (mloc < valid) {
                const int   t  = tok [e * LISTSTRIDE + mb + mloc];
                const float gw = gate[e * LISTSTRIDE + mb + mloc];
                const float bv = bias[e * LISTSTRIDE + mb + mloc];
                float* op = out + (long long)t * NN + nb;
                #pragma unroll
                for (int nt = 0; nt < 4; ++nt)
                    atomicAdd(op + nt * 16 + laneN, gw * (acc[mi][nt][i] + bv));
            }
        }
    }
}

// ---------------------------------------------------------------- launch
extern "C" void kernel_launch(void* const* d_in, const int* in_sizes, int n_in,
                              void* d_out, int out_size, void* d_ws, size_t ws_size,
                              hipStream_t stream) {
    const float* x      = (const float*)d_in[0];
    const int*   w_q    = (const int*)  d_in[1];
    const float* scales = (const float*)d_in[2];
    const float* zeros  = (const float*)d_in[3];
    const float* topk_w = (const float*)d_in[4];
    const int*   topk_i = (const int*)  d_in[5];
    float* out = (float*)d_out;

    // workspace layout (4-byte units): cnt[pad 64] | tok[E*2048] | gate[E*2048] | bias[E*2048]
    int*   cnt  = (int*)d_ws;
    int*   tok  = cnt + 64;
    float* gate = (float*)(tok + EXP * LISTSTRIDE);
    float* bias = gate + EXP * LISTSTRIDE;

    moe_init_kernel<<<(TT * NN + 255) / 256, 256, 0, stream>>>(out, cnt);
    moe_route_kernel<<<(TT * TOPK + 255) / 256, 256, 0, stream>>>(topk_i, topk_w, cnt, tok, gate);
    moe_bias_kernel<<<EXP * 256, 256, 0, stream>>>(x, zeros, tok, cnt, bias);
    dim3 grid(NN / BLOCK_N, (TT * TOPK) / BLOCK_M, EXP);
    moe_gemm_kernel<<<grid, 256, 0, stream>>>(x, w_q, scales, tok, gate, bias, cnt, out);
}
